// TransformerDecoder_47974784696876
// MI455X (gfx1250) — matrix operand order repaired and checked by reading, not compiled, not run on hardware
//
#include <hip/hip_runtime.h>
#include <math.h>

// ---------------------------------------------------------------------------
// Problem constants (from the reference): B=4, S=1024, E=1024, D=1024, H=16,
// L=6, F=4096, DPH=64. All GEMM dims are multiples of 128.
// ---------------------------------------------------------------------------
#define CB   4
#define CS   1024
#define CE   1024
#define CD   1024
#define CH   16
#define CL   6
#define CF   4096
#define CDPH 64
#define NEGF (-1e30f)

typedef __attribute__((ext_vector_type(16))) _Float16 v16h;
typedef __attribute__((ext_vector_type(8)))  float    v8f;

union V16 { v16h h; uint4 u[2]; };

static __device__ __forceinline__ v8f wmma_f16(v16h a, v16h b, v8f c) {
  return __builtin_amdgcn_wmma_f32_16x16x32_f16(false, a, false, b, (short)0, c,
                                                false, false);
}

static __device__ __forceinline__ v8f v8f_zero() {
  v8f z = {0.f, 0.f, 0.f, 0.f, 0.f, 0.f, 0.f, 0.f};
  return z;
}

// ---------------------------------------------------------------------------
// GEMM:  C[M,N] = A[M,K] (f32) x W[K,N] (f32) + bias[N], optional exact GELU.
// f32 -> f16 conversion into LDS; v_wmma_f32_16x16x32_f16 with f32 accum.
// Block: 256 threads (8 waves). Tile 128x128, K-step 32.
// Wave (wm,wn) in 4x2 grid computes 32x64 = 2x4 WMMA tiles.
// ---------------------------------------------------------------------------
__global__ __launch_bounds__(256) void gemm_bias(
    const float* __restrict__ A, const float* __restrict__ W,
    const float* __restrict__ bias, float* __restrict__ C,
    int M, int N, int K, int act) {
  __shared__ __align__(16) _Float16 As[128][40];  // [m][k], pad 8
  __shared__ __align__(16) _Float16 Bs[128][40];  // [n][k] (transposed), pad 8

  const int tid  = threadIdx.x;
  const int lane = tid & 31;
  const int wid  = tid >> 5;
  const int wm   = wid & 3;   // 0..3 -> 32-row strip
  const int wn   = wid >> 2;  // 0..1 -> 64-col strip
  const int bm   = blockIdx.x * 128;
  const int bn   = blockIdx.y * 128;

  v8f acc[2][4];
#pragma unroll
  for (int i = 0; i < 2; ++i)
#pragma unroll
    for (int j = 0; j < 4; ++j) acc[i][j] = v8f_zero();

  const int arow = tid >> 1, acol = (tid & 1) * 16;   // 128 rows x 32 cols
  const int brow = tid >> 3, bcol = (tid & 7) * 16;   // 32 rows  x 128 cols
  const float* aptr = A + (size_t)(bm + arow) * K + acol;
  const float* bptr = W + (size_t)brow * N + bn + bcol;

  for (int k0 = 0; k0 < K; k0 += 32) {
#pragma unroll
    for (int i = 0; i < 16; i += 4) {
      float4 fa = *(const float4*)(aptr + k0 + i);
      As[arow][acol + i + 0] = (_Float16)fa.x;
      As[arow][acol + i + 1] = (_Float16)fa.y;
      As[arow][acol + i + 2] = (_Float16)fa.z;
      As[arow][acol + i + 3] = (_Float16)fa.w;
      float4 fb = *(const float4*)(bptr + (size_t)k0 * N + i);
      Bs[bcol + i + 0][brow] = (_Float16)fb.x;
      Bs[bcol + i + 1][brow] = (_Float16)fb.y;
      Bs[bcol + i + 2][brow] = (_Float16)fb.z;
      Bs[bcol + i + 3][brow] = (_Float16)fb.w;
    }
    if (k0 + 32 < K) {  // prefetch next K tile -> global_prefetch_b8
      __builtin_prefetch(aptr + k0 + 32, 0, 1);
      __builtin_prefetch(bptr + (size_t)(k0 + 32) * N, 0, 1);
    }
    __syncthreads();

    // A fragments: lane m=lane%15-ish; per ISA 16-bit A layout:
    // base = (lane<16 ? 0 : 8); elems 0..7 -> K=base..base+7,
    // elems 8..15 -> K=base+16..base+23. Two b128 LDS reads.
    V16 af[2];
#pragma unroll
    for (int mt = 0; mt < 2; ++mt) {
      const int m  = wm * 32 + mt * 16 + (lane & 15);
      const int kb = (lane < 16) ? 0 : 8;
      af[mt].u[0] = *(const uint4*)&As[m][kb];
      af[mt].u[1] = *(const uint4*)&As[m][kb + 16];
    }
#pragma unroll
    for (int nt = 0; nt < 4; ++nt) {
      // B fragment: n = lane%16; same K order as A: base = (lane<16 ? 0 : 8),
      // elems 0..7 -> K=base..base+7, elems 8..15 -> K=base+16..base+23.
      const int n  = wn * 64 + nt * 16 + (lane & 15);
      const int kb = (lane < 16) ? 0 : 8;
      V16 bf;
      bf.u[0] = *(const uint4*)&Bs[n][kb];
      bf.u[1] = *(const uint4*)&Bs[n][kb + 16];
#pragma unroll
      for (int mt = 0; mt < 2; ++mt)
        acc[mt][nt] = wmma_f16(af[mt].h, bf.h, acc[mt][nt]);
    }
    __syncthreads();
  }

  // Epilogue: C layout: VGPR r -> row (lane<16 ? r : r+8), col = lane%16.
#pragma unroll
  for (int mt = 0; mt < 2; ++mt)
#pragma unroll
    for (int nt = 0; nt < 4; ++nt) {
      const int   col = bn + wn * 64 + nt * 16 + (lane & 15);
      const float bv  = bias[col];
#pragma unroll
      for (int r = 0; r < 8; ++r) {
        const int row = bm + wm * 32 + mt * 16 + ((lane < 16) ? r : r + 8);
        float v = acc[mt][nt][r] + bv;
        if (act == 1) v = 0.5f * v * (1.0f + erff(v * 0.70710678118654752f));
        C[(size_t)row * N + col] = v;
      }
    }
}

// ---------------------------------------------------------------------------
// Flash attention (WMMA). Block = 128 threads (4 waves); each wave owns a
// 16-row Q tile, block covers 64 q rows of one (b,h). Keys processed 32/iter:
// S16x32 = Q(16x64)K^T via 4 wmma, online softmax, P bounced through LDS to
// A-fragment layout, O(16x64) += P(16x32)V(32x64) via 4 wmma.
// Strided addressing covers both self-attn (qkv[...,c], dim stride 3) and
// cross-attn (encoder_state layout).
// ---------------------------------------------------------------------------
__global__ __launch_bounds__(128) void flash_attn(
    const float* __restrict__ Qp, long qB, long qS, long qH, long qD,
    const float* __restrict__ Kp, const float* __restrict__ Vp,
    long kB, long kS, long kH, long kD,
    float* __restrict__ Out, int nkeys, int causal,
    const unsigned char* __restrict__ kmask, int mask_stride, float scale) {
  __shared__ __align__(16) _Float16 Kt[32][72];     // [key][dim]
  __shared__ __align__(16) _Float16 Vt[64][40];     // [dim][key]
  __shared__ __align__(16) _Float16 Pt[4][16][40];  // per-wave P (row, key)

  const int tid   = threadIdx.x;
  const int lane  = tid & 31;
  const int wave  = tid >> 5;
  const int b     = blockIdx.z;
  const int h     = blockIdx.y;
  const int qblk  = blockIdx.x * 64;
  const int qbase = qblk + wave * 16;

  // Resident Q fragments (two K=32 chunks covering DPH=64), scaled by 1/sqrt(DPH).
  V16 qf[2];
  {
    const int m  = lane & 15;
    const int kb = (lane < 16) ? 0 : 8;
    const float* qrow =
        Qp + (size_t)b * qB + (size_t)(qbase + m) * qS + (size_t)h * qH;
#pragma unroll
    for (int kc = 0; kc < 2; ++kc)
#pragma unroll
      for (int e = 0; e < 16; ++e) {
        const int d = kc * 32 + kb + ((e < 8) ? e : e + 8);
        qf[kc].h[e] = (_Float16)(qrow[(size_t)d * qD] * scale);
      }
  }

  float rm[8], rl[8];
#pragma unroll
  for (int r = 0; r < 8; ++r) { rm[r] = NEGF; rl[r] = 0.f; }
  v8f o[4];
#pragma unroll
  for (int dtile = 0; dtile < 4; ++dtile) o[dtile] = v8f_zero();

  const int skey = tid & 31;           // staging: key within 32-block
  const int sdg  = (tid >> 5) * 16;    // staging: 16-dim group
  const int jend = causal ? ((qblk + 64 < nkeys) ? qblk + 64 : nkeys) : nkeys;

  for (int j0 = 0; j0 < jend; j0 += 32) {
    {  // cooperative K/V staging (f32 -> f16)
      const size_t kvoff =
          (size_t)b * kB + (size_t)(j0 + skey) * kS + (size_t)h * kH;
      const float* kr = Kp + kvoff;
      const float* vr = Vp + kvoff;
#pragma unroll
      for (int e = 0; e < 16; ++e) {
        Kt[skey][sdg + e] = (_Float16)kr[(size_t)(sdg + e) * kD];
        Vt[sdg + e][skey] = (_Float16)vr[(size_t)(sdg + e) * kD];
      }
    }
    __syncthreads();

    // Scores: two 16x16 N-tiles x two K=32 chunks.
    v8f sc[2];
    sc[0] = v8f_zero();
    sc[1] = v8f_zero();
#pragma unroll
    for (int nt = 0; nt < 2; ++nt) {
      const int kl  = nt * 16 + (lane & 15);   // key (B-matrix column)
      const int kb2 = (lane < 16) ? 0 : 8;     // contraction base
#pragma unroll
      for (int kc = 0; kc < 2; ++kc) {
        V16 bf;
        bf.u[0] = *(const uint4*)&Kt[kl][kc * 32 + kb2];
        bf.u[1] = *(const uint4*)&Kt[kl][kc * 32 + kb2 + 16];
        sc[nt] = wmma_f16(qf[kc].h, bf.h, sc[nt]);
      }
    }

    // Online softmax: each C-layout row lives in 16 lanes of one half-wave.
    const int cb = j0 + (lane & 15);
#pragma unroll
    for (int r = 0; r < 8; ++r) {
      const int row = qbase + ((lane < 16) ? r : r + 8);
      float s0 = sc[0][r], s1 = sc[1][r];
      const int c0 = cb, c1 = cb + 16;
      const bool m0 = (c0 < nkeys) && (!causal || c0 <= row) &&
                      kmask[(size_t)b * mask_stride + c0];
      const bool m1 = (c1 < nkeys) && (!causal || c1 <= row) &&
                      kmask[(size_t)b * mask_stride + c1];
      s0 = m0 ? s0 : NEGF;
      s1 = m1 ? s1 : NEGF;
      float mx = fmaxf(s0, s1);
      mx = fmaxf(mx, __shfl_xor(mx, 1, 32));
      mx = fmaxf(mx, __shfl_xor(mx, 2, 32));
      mx = fmaxf(mx, __shfl_xor(mx, 4, 32));
      mx = fmaxf(mx, __shfl_xor(mx, 8, 32));
      const float mnew  = fmaxf(rm[r], mx);
      const float alpha = __expf(rm[r] - mnew);
      const float p0    = m0 ? __expf(s0 - mnew) : 0.f;
      const float p1    = m1 ? __expf(s1 - mnew) : 0.f;
      float ps = p0 + p1;
      ps += __shfl_xor(ps, 1, 32);
      ps += __shfl_xor(ps, 2, 32);
      ps += __shfl_xor(ps, 4, 32);
      ps += __shfl_xor(ps, 8, 32);
      rl[r] = rl[r] * alpha + ps;
      rm[r] = mnew;
#pragma unroll
      for (int dtile = 0; dtile < 4; ++dtile) o[dtile][r] *= alpha;
      const int prow = (lane < 16) ? r : r + 8;
      Pt[wave][prow][(lane & 15)]      = (_Float16)p0;
      Pt[wave][prow][(lane & 15) + 16] = (_Float16)p1;
    }
    // Same-wave LDS store->load reshape; DS ops are in-order per wave, the
    // explicit wait is belt-and-braces.
    asm volatile("s_wait_dscnt 0x0" ::: "memory");

    // P as A-fragment (16x32, K = local key index).
    V16 pf;
    {
      const int m  = lane & 15;
      const int kb = (lane < 16) ? 0 : 8;
      pf.u[0] = *(const uint4*)&Pt[wave][m][kb];
      pf.u[1] = *(const uint4*)&Pt[wave][m][kb + 16];
    }
    // O += P @ V : four 16-dim output tiles.
#pragma unroll
    for (int dtile = 0; dtile < 4; ++dtile) {
      const int dcol = dtile * 16 + (lane & 15);
      const int kb2  = (lane < 16) ? 0 : 8;
      V16 bf;
      bf.u[0] = *(const uint4*)&Vt[dcol][kb2];
      bf.u[1] = *(const uint4*)&Vt[dcol][kb2 + 16];
      o[dtile] = wmma_f16(pf.h, bf.h, o[dtile]);
    }
    __syncthreads();
  }

  // Normalize and write (B,S,D) with head offset.
#pragma unroll
  for (int dtile = 0; dtile < 4; ++dtile) {
    const int d = dtile * 16 + (lane & 15);
#pragma unroll
    for (int r = 0; r < 8; ++r) {
      const int   row   = qbase + ((lane < 16) ? r : r + 8);
      const float denom = rl[r];
      const float val   = (denom > 0.f) ? (o[dtile][r] / denom) : 0.f;
      Out[((size_t)b * CS + row) * CD + h * CDPH + d] = val;
    }
  }
}

// ---------------------------------------------------------------------------
// Fused residual add + LayerNorm over D=1024. One block per row; 256 threads,
// 4 elements each; wave shfl reduction then cross-wave LDS reduction.
// Safe in-place (X == Out): all reads of X precede the barrier, writes follow.
// ---------------------------------------------------------------------------
__global__ __launch_bounds__(256) void add_ln(
    const float* __restrict__ X, const float* __restrict__ R,
    const float* __restrict__ G, const float* __restrict__ Bb,
    float* __restrict__ Out) {
  const int row = blockIdx.x;
  const int tid = threadIdx.x;
  const float* xr = X + (size_t)row * CD;
  const float* rr = R + (size_t)row * CD;
  const int c = tid * 4;
  float4 xv = *(const float4*)(xr + c);
  float4 rv = *(const float4*)(rr + c);
  const float v0 = xv.x + rv.x, v1 = xv.y + rv.y;
  const float v2 = xv.z + rv.z, v3 = xv.w + rv.w;
  float s  = v0 + v1 + v2 + v3;
  float s2 = v0 * v0 + v1 * v1 + v2 * v2 + v3 * v3;
#pragma unroll
  for (int off = 16; off >= 1; off >>= 1) {
    s  += __shfl_xor(s, off, 32);
    s2 += __shfl_xor(s2, off, 32);
  }
  __shared__ float ws[8], ws2[8];
  if ((tid & 31) == 0) { ws[tid >> 5] = s; ws2[tid >> 5] = s2; }
  __syncthreads();
  float S = 0.f, S2 = 0.f;
#pragma unroll
  for (int w = 0; w < 8; ++w) { S += ws[w]; S2 += ws2[w]; }
  const float mean = S * (1.0f / CD);
  const float var  = S2 * (1.0f / CD) - mean * mean;
  const float inv  = rsqrtf(var + 1e-5f);
  float4 g  = *(const float4*)(G + c);
  float4 bb = *(const float4*)(Bb + c);
  float4 ov;
  ov.x = (v0 - mean) * inv * g.x + bb.x;
  ov.y = (v1 - mean) * inv * g.y + bb.y;
  ov.z = (v2 - mean) * inv * g.z + bb.z;
  ov.w = (v3 - mean) * inv * g.w + bb.w;
  *(float4*)(Out + (size_t)row * CD + c) = ov;
}

// ---------------------------------------------------------------------------
// Driver: 6 layers of [QKV gemm -> self flash-attn -> out-proj -> add+LN ->
// cross-q gemm -> cross flash-attn -> out-proj -> add+LN -> FFN (gelu) ->
// add+LN]. Last LN writes directly to d_out.
// ---------------------------------------------------------------------------
extern "C" void kernel_launch(void* const* d_in, const int* in_sizes, int n_in,
                              void* d_out, int out_size, void* d_ws,
                              size_t ws_size, hipStream_t stream) {
  (void)in_sizes; (void)n_in; (void)out_size; (void)ws_size;
  const float* tensor = (const float*)d_in[0];
  const float* enc    = (const float*)d_in[1];
  const float* w_qkv  = (const float*)d_in[2];
  const float* b_qkv  = (const float*)d_in[3];
  const float* w_so   = (const float*)d_in[4];
  const float* b_so   = (const float*)d_in[5];
  const float* w_cq   = (const float*)d_in[6];
  const float* b_cq   = (const float*)d_in[7];
  const float* w_co   = (const float*)d_in[8];
  const float* b_co   = (const float*)d_in[9];
  const float* w_f1   = (const float*)d_in[10];
  const float* b_f1   = (const float*)d_in[11];
  const float* w_f2   = (const float*)d_in[12];
  const float* b_f2   = (const float*)d_in[13];
  const float* ln1g   = (const float*)d_in[14];
  const float* ln1b   = (const float*)d_in[15];
  const float* ln2g   = (const float*)d_in[16];
  const float* ln2b   = (const float*)d_in[17];
  const float* ln3g   = (const float*)d_in[18];
  const float* ln3b   = (const float*)d_in[19];
  const unsigned char* dmask = (const unsigned char*)d_in[20];
  const unsigned char* emask = (const unsigned char*)d_in[21];
  float* out = (float*)d_out;

  const size_t nBSD = (size_t)CB * CS * CD;  // 4M floats
  float* ws  = (float*)d_ws;
  float* x   = ws;                              // 16 MB
  float* qkv = x + nBSD;                        // 48 MB
  float* t0  = qkv + (size_t)CB * CS * 3 * CD;  // 16 MB
  float* t1  = t0 + nBSD;                       // 16 MB
  float* q2  = t1 + nBSD;                       // 16 MB
  float* hb  = q2 + nBSD;                       // 64 MB (B*S*F)

  hipMemcpyAsync(x, tensor, nBSD * sizeof(float), hipMemcpyDeviceToDevice,
                 stream);

  const int M = CB * CS;  // 4096
  const float scale = 1.0f / 8.0f;  // 1/sqrt(DPH)
  const dim3 gQKV(M / 128, (3 * CD) / 128);
  const dim3 gDD(M / 128, CD / 128);
  const dim3 gDF(M / 128, CF / 128);
  const dim3 gAttn(CS / 64, CH, CB);

  for (int l = 0; l < CL; ++l) {
    // 1) QKV projection
    gemm_bias<<<gQKV, 256, 0, stream>>>(x, w_qkv + (size_t)l * CD * 3 * CD,
                                        b_qkv + (size_t)l * 3 * CD, qkv, M,
                                        3 * CD, CD, 0);
    // 2) Causal self-attention: q/k/v interleaved with element stride 3
    flash_attn<<<gAttn, 128, 0, stream>>>(
        qkv + 0, (long)CS * 3 * CD, (long)3 * CD, (long)CDPH * 3, 3L,
        qkv + 1, qkv + 2,
        (long)CS * 3 * CD, (long)3 * CD, (long)CDPH * 3, 3L,
        t0, CS, 1, dmask, CS, scale);
    // 3) Self-attention output projection
    gemm_bias<<<gDD, 256, 0, stream>>>(t0, w_so + (size_t)l * CD * CD,
                                       b_so + (size_t)l * CD, t1, M, CD, CD, 0);
    // 4) x = LN1(x + sa)
    add_ln<<<M, 256, 0, stream>>>(x, t1, ln1g + (size_t)l * CD,
                                  ln1b + (size_t)l * CD, x);
    // 5) Cross query
    gemm_bias<<<gDD, 256, 0, stream>>>(x, w_cq + (size_t)l * CD * CD,
                                       b_cq + (size_t)l * CD, q2, M, CD, CD, 0);
    // 6) Cross attention: K/V from encoder_state[:, :, l, {0,1}]
    flash_attn<<<gAttn, 128, 0, stream>>>(
        q2, (long)CS * CD, (long)CD, (long)CDPH, 1L,
        enc + (size_t)(2 * l + 0) * CE * CDPH,
        enc + (size_t)(2 * l + 1) * CE * CDPH,
        (long)CH * CL * 2 * CE * CDPH, (long)CDPH, (long)CL * 2 * CE * CDPH, 1L,
        t0, CE, 0, emask, CE, scale);
    // 7) Cross-attention output projection
    gemm_bias<<<gDD, 256, 0, stream>>>(t0, w_co + (size_t)l * CD * CD,
                                       b_co + (size_t)l * CD, t1, M, CD, CD, 0);
    // 8) x = LN2(x + co)
    add_ln<<<M, 256, 0, stream>>>(x, t1, ln2g + (size_t)l * CD,
                                  ln2b + (size_t)l * CD, x);
    // 9) FFN up-projection with fused exact GELU
    gemm_bias<<<gDF, 256, 0, stream>>>(x, w_f1 + (size_t)l * CD * CF,
                                       b_f1 + (size_t)l * CF, hb, M, CF, CD, 1);
    // 10) FFN down-projection
    gemm_bias<<<gDD, 256, 0, stream>>>(hb, w_f2 + (size_t)l * CF * CD,
                                       b_f2 + (size_t)l * CD, t1, M, CD, CF, 0);
    // 11) x = LN3(x + ffn); final layer writes d_out directly
    float* lnout = (l == CL - 1) ? out : x;
    add_ln<<<M, 256, 0, stream>>>(x, t1, ln3g + (size_t)l * CD,
                                  ln3b + (size_t)l * CD, lnout);
  }
}
